// SequenceEncoder_34325378629899
// MI455X (gfx1250) — compile-verified, matrix operand order checked by reading, not run on hardware
//
#include <hip/hip_runtime.h>
#include <hip/hip_bf16.h>

// ---------------------------------------------------------------------------
// SequenceEncoder forward for MI455X (gfx1250)
//   wave32, WMMA f32_16x16x32_f16, double-buffered async global->LDS staging
// B=4, S=2048, D=1024, H=16, hd=64, L=6, VOCAB=21, SEQ_DIM=128
// ---------------------------------------------------------------------------

#define B_   4
#define S_   2048
#define D_   1024
#define H_   16
#define HD_  64
#define L_   6
#define SEQDIM_ 128
#define NTOK (B_ * S_)        // 8192
#define EPS_ 1e-5f

typedef __attribute__((ext_vector_type(16))) _Float16 v16h;
typedef __attribute__((ext_vector_type(8)))  float    v8f;

__device__ __forceinline__ v8f wmma_f16(v16h a, v16h b, v8f c) {
  return __builtin_amdgcn_wmma_f32_16x16x32_f16(
      /*neg_a=*/false, a, /*neg_b=*/false, b,
      /*c_mod=*/(short)0, c, /*reuse_a=*/false, /*reuse_b=*/false);
}

// Fragment load (A or B operand).  p -> start of this lane's 32-element K-run.
// CDNA5 16-bit layout: lane holds K = {kb..kb+7, 16+kb..16+kb+7},
// kb = 0 (lanes 0-15) / 8 (lanes 16-31).  Two contiguous 16B loads.
__device__ __forceinline__ v16h frag_ld(const _Float16* p) {
  int lane = threadIdx.x & 31;
  int kb = (lane & 16) ? 8 : 0;
  union { v16h v; uint4 u[2]; } f;
  f.u[0] = *(const uint4*)(p + kb);
  f.u[1] = *(const uint4*)(p + 16 + kb);
  return f.v;
}

// Async copy 16B from global to LDS (CDNA5 async path, tracked by ASYNCcnt).
__device__ __forceinline__ void async_ld_b128(unsigned lds_off, const void* g) {
  asm volatile("global_load_async_to_lds_b128 %0, %1, off"
               :: "v"(lds_off), "v"((unsigned long long)(uintptr_t)g)
               : "memory");
}
__device__ __forceinline__ void wait_async_le4() {
  asm volatile("s_wait_asynccnt 0x4" ::: "memory");
}
__device__ __forceinline__ void wait_async0() {
  asm volatile("s_wait_asynccnt 0x0" ::: "memory");
}

// ---------------------------------------------------------------------------
// One-time weight repack: Wt[n][k] = (f16) W[k][n].   K = 1<<kbits.
// ---------------------------------------------------------------------------
__global__ __launch_bounds__(256) void pack_wt(const float* __restrict__ W,
                                               _Float16* __restrict__ Wt,
                                               int kbits, int N) {
  size_t idx = (size_t)blockIdx.x * 256 + threadIdx.x;   // over N*K, k fastest
  int n = (int)(idx >> kbits);
  int k = (int)(idx & ((1u << kbits) - 1));
  Wt[idx] = (_Float16)W[(size_t)k * N + n];
}

// ---------------------------------------------------------------------------
// Embedding gather -> f16:  xemb16[t, c] = emb[seq[t], c]
// ---------------------------------------------------------------------------
__global__ __launch_bounds__(128) void gather_embed(const int* __restrict__ seq,
                                                    const float* __restrict__ emb,
                                                    _Float16* __restrict__ out) {
  int t = blockIdx.x;
  int c = threadIdx.x;
  out[(size_t)t * SEQDIM_ + c] = (_Float16)emb[(size_t)seq[t] * SEQDIM_ + c];
}

// ---------------------------------------------------------------------------
// WMMA GEMM:  C[M,N] = A[M,K] (f16) @ Wt[N,K]^T (f16) + bias
// 256 threads / 8 waves, 128x128 tile, BK=32; wave = 32x64 -> 8 accumulators.
// Double-buffered LDS; tiles staged with global_load_async_to_lds_b128 and
// drained with s_wait_asynccnt (in-order completion -> "<=4" = prev tile done).
// Epilogue MODE: 0 = f32 row-major          (o-proj)
//               1 = f32 + f16 row-major     (input proj -> x, x16)
//               2 = f16 [B,H,S,hd]          (Q/K)
//               3 = f16 [B,H,hd,S]          (V transposed)
// ---------------------------------------------------------------------------
#define LDSTR 40   // padded LDS row stride in halves (80B, 16B aligned)

template <int MODE>
__global__ __launch_bounds__(256) void gemm_wmma(const _Float16* __restrict__ A,
                                                 const _Float16* __restrict__ Wt,
                                                 const float* __restrict__ bias,
                                                 float* __restrict__ Cf,
                                                 _Float16* __restrict__ Ch,
                                                 int M, int N, int K) {
  __shared__ _Float16 As[2][128 * LDSTR];
  __shared__ _Float16 Bs[2][128 * LDSTR];
  unsigned asB0 = (unsigned)(uintptr_t)&As[0][0];
  unsigned asB1 = (unsigned)(uintptr_t)&As[1][0];
  unsigned bsB0 = (unsigned)(uintptr_t)&Bs[0][0];
  unsigned bsB1 = (unsigned)(uintptr_t)&Bs[1][0];

  int bm = blockIdx.x * 128;
  int bn = blockIdx.y * 128;
  int tid  = threadIdx.x;
  int wave = tid >> 5;
  int lane = tid & 31;
  int lrow = lane & 15;
  int wm = (wave >> 1) * 32;   // 0,32,64,96
  int wn = (wave & 1) * 64;    // 0,64

  // this thread's two 16B staging chunks (512 chunks per 128x32 tile)
  int row0 = tid >> 2;                 // 0..63
  int c0   = (tid & 3) * 8;
  int row1 = row0 + 64;                // 64..127
  int c1   = c0;
  unsigned l0 = (unsigned)(row0 * (LDSTR * 2) + c0 * 2);
  unsigned l1 = (unsigned)(row1 * (LDSTR * 2) + c1 * 2);
  const _Float16* Arow0 = A  + (size_t)(bm + row0) * K + c0;
  const _Float16* Arow1 = A  + (size_t)(bm + row1) * K + c1;
  const _Float16* Brow0 = Wt + (size_t)(bn + row0) * K + c0;
  const _Float16* Brow1 = Wt + (size_t)(bn + row1) * K + c1;

  v8f acc[2][4] = {};

  // prologue: stage tile 0 into buffer 0
  async_ld_b128(asB0 + l0, Arow0);
  async_ld_b128(bsB0 + l0, Brow0);
  async_ld_b128(asB0 + l1, Arow1);
  async_ld_b128(bsB0 + l1, Brow1);

  int nk = K >> 5;
  for (int i = 0; i < nk; ++i) {
    int p = i & 1;
    if (i + 1 < nk) {
      int kn = (i + 1) * 32;
      unsigned aB = p ? asB0 : asB1;
      unsigned bB = p ? bsB0 : bsB1;
      async_ld_b128(aB + l0, Arow0 + kn);
      async_ld_b128(bB + l0, Brow0 + kn);
      async_ld_b128(aB + l1, Arow1 + kn);
      async_ld_b128(bB + l1, Brow1 + kn);
      wait_async_le4();     // in-order: tile i finished landing
    } else {
      wait_async0();
    }
    __syncthreads();

    const _Float16* Ap = As[p];
    const _Float16* Bp = Bs[p];
    v16h a0 = frag_ld(Ap + (wm + lrow) * LDSTR);
    v16h a1 = frag_ld(Ap + (wm + 16 + lrow) * LDSTR);
#pragma unroll
    for (int j = 0; j < 4; ++j) {
      v16h bj = frag_ld(Bp + (wn + j * 16 + lrow) * LDSTR);
      acc[0][j] = wmma_f16(a0, bj, acc[0][j]);
      acc[1][j] = wmma_f16(a1, bj, acc[1][j]);
    }
    __syncthreads();   // tile p fully consumed; safe to overwrite next round
  }

  // Epilogue.  C layout: lane n = lane&15; reg r -> row r (+8 for lanes>=16).
#pragma unroll
  for (int i = 0; i < 2; ++i) {
#pragma unroll
    for (int j = 0; j < 4; ++j) {
      int n  = bn + wn + j * 16 + lrow;
      int mb = bm + wm + i * 16 + ((lane & 16) ? 8 : 0);
      float bb = bias[n];
#pragma unroll
      for (int r = 0; r < 8; ++r) {
        float v = acc[i][j][r] + bb;
        int m = mb + r;
        if (MODE == 0) {
          Cf[(size_t)m * N + n] = v;
        } else if (MODE == 1) {
          Cf[(size_t)m * N + n] = v;
          Ch[(size_t)m * N + n] = (_Float16)v;
        } else if (MODE == 2) {
          int b = m >> 11, s = m & 2047, h = n >> 6, d = n & 63;
          Ch[((((size_t)b * H_ + h) * S_ + s) << 6) + d] = (_Float16)v;
        } else {
          int b = m >> 11, s = m & 2047, h = n >> 6, d = n & 63;
          Ch[((((size_t)b * H_ + h) * HD_ + d) << 11) + s] = (_Float16)v;
        }
      }
    }
  }
}

// ---------------------------------------------------------------------------
// Flash attention: grid (S/64, B*H), 128 threads = 4 waves, 16 q-rows/wave.
// 64-key chunks: 16 WMMA per chunk; softmax shuffle/exp overhead amortized
// over twice as many keys as a 32-key pipeline.  Writes f16 [NTOK, D].
// ---------------------------------------------------------------------------
#define PSTR 72    // Pls row stride in halves (144B, 16B aligned)

__global__ __launch_bounds__(128) void flash_attn(const _Float16* __restrict__ Q,
                                                  const _Float16* __restrict__ K,
                                                  const _Float16* __restrict__ Vt,
                                                  const int* __restrict__ mask,
                                                  _Float16* __restrict__ O16,
                                                  float scale) {
  __shared__ _Float16 Pls[4][16 * PSTR];   // per-wave 16x64 P scratch

  int wave = threadIdx.x >> 5;
  int lane = threadIdx.x & 31;
  int lrow = lane & 15;
  int bh = blockIdx.y;
  int b  = bh >> 4;
  int h  = bh & 15;
  int q0 = blockIdx.x * 64 + wave * 16;

  const _Float16* Qh = Q  + (size_t)bh * S_ * HD_;
  const _Float16* Kh = K  + (size_t)bh * S_ * HD_;
  const _Float16* Vh = Vt + (size_t)bh * HD_ * S_;
  const int* mrow = mask + (size_t)b * S_;

  v16h qa0 = frag_ld(Qh + (size_t)(q0 + lrow) * HD_);
  v16h qa1 = frag_ld(Qh + (size_t)(q0 + lrow) * HD_ + 32);

  v8f o[4] = {};
  float mrun[8], lrun[8];
#pragma unroll
  for (int r = 0; r < 8; ++r) { mrun[r] = -1e30f; lrun[r] = 0.f; }

  for (int j = 0; j < S_; j += 64) {
    // ---- scores: 4 n-tiles x (K=64 as 2 chunks) = 8 WMMA ----
    v8f s[4];
#pragma unroll
    for (int t = 0; t < 4; ++t) {
      int key = j + t * 16 + lrow;
      v16h b0 = frag_ld(Kh + (size_t)key * HD_);
      v16h b1 = frag_ld(Kh + (size_t)key * HD_ + 32);
      v8f z = {};
      z = wmma_f16(qa0, b0, z);
      z = wmma_f16(qa1, b1, z);
      bool dead = (mrow[key] == 0);
#pragma unroll
      for (int r = 0; r < 8; ++r)
        s[t][r] = dead ? -1e9f : z[r] * scale;
    }

    // ---- online softmax over 64 keys ----
    float mn[8], corr[8];
#pragma unroll
    for (int r = 0; r < 8; ++r) {
      float v = fmaxf(fmaxf(s[0][r], s[1][r]), fmaxf(s[2][r], s[3][r]));
#pragma unroll
      for (int off = 1; off < 16; off <<= 1)
        v = fmaxf(v, __shfl_xor(v, off, 32));
      mn[r]   = fmaxf(mrun[r], v);
      corr[r] = __expf(mrun[r] - mn[r]);
      mrun[r] = mn[r];
    }

    float p[4][8];
#pragma unroll
    for (int r = 0; r < 8; ++r) {
#pragma unroll
      for (int t = 0; t < 4; ++t)
        p[t][r] = __expf(s[t][r] - mn[r]);
      float v = (p[0][r] + p[1][r]) + (p[2][r] + p[3][r]);
#pragma unroll
      for (int off = 1; off < 16; off <<= 1)
        v += __shfl_xor(v, off, 32);
      lrun[r] = lrun[r] * corr[r] + v;
    }

#pragma unroll
    for (int nt = 0; nt < 4; ++nt)
#pragma unroll
      for (int r = 0; r < 8; ++r)
        o[nt][r] *= corr[r];

    // ---- transpose P (C-layout) -> A-layout via per-wave LDS scratch ----
    __syncthreads();
    {
      int prow = (lane & 16) ? 8 : 0;
#pragma unroll
      for (int r = 0; r < 8; ++r)
#pragma unroll
        for (int t = 0; t < 4; ++t)
          Pls[wave][(prow + r) * PSTR + t * 16 + lrow] = (_Float16)p[t][r];
    }
    __syncthreads();
    v16h pa0 = frag_ld(&Pls[wave][lrow * PSTR]);
    v16h pa1 = frag_ld(&Pls[wave][lrow * PSTR + 32]);

    // ---- O += P @ V : 4 d-tiles x 2 k-chunks = 8 WMMA ----
#pragma unroll
    for (int nt = 0; nt < 4; ++nt) {
      const _Float16* vp = Vh + (size_t)(nt * 16 + lrow) * S_ + j;
      v16h bv0 = frag_ld(vp);
      v16h bv1 = frag_ld(vp + 32);
      o[nt] = wmma_f16(pa0, bv0, o[nt]);
      o[nt] = wmma_f16(pa1, bv1, o[nt]);
    }
  }

  int mb = q0 + ((lane & 16) ? 8 : 0);
#pragma unroll
  for (int nt = 0; nt < 4; ++nt) {
    int d = nt * 16 + lrow;
#pragma unroll
    for (int r = 0; r < 8; ++r) {
      float inv = 1.0f / lrun[r];
      O16[((size_t)b * S_ + mb + r) * D_ + h * HD_ + d] =
          (_Float16)(o[nt][r] * inv);
    }
  }
}

// ---------------------------------------------------------------------------
// Residual + LayerNorm; writes f32 stream and the f16 copy for next layer.
// ---------------------------------------------------------------------------
__global__ __launch_bounds__(256) void add_layernorm(const float* __restrict__ x,
                                                     const float* __restrict__ y,
                                                     const float* __restrict__ g,
                                                     const float* __restrict__ beta,
                                                     float* __restrict__ dstf,
                                                     _Float16* __restrict__ dsth) {
  __shared__ float red[256];
  int row = blockIdx.x;
  int tid = threadIdx.x;
  const float* xr = x + (size_t)row * D_;
  const float* yr = y + (size_t)row * D_;

  float v[4];
  float sum = 0.f;
#pragma unroll
  for (int i = 0; i < 4; ++i) {
    int c = tid + i * 256;
    v[i] = xr[c] + yr[c];
    sum += v[i];
  }
  red[tid] = sum; __syncthreads();
  for (int sft = 128; sft > 0; sft >>= 1) {
    if (tid < sft) red[tid] += red[tid + sft];
    __syncthreads();
  }
  float mu = red[0] * (1.0f / D_);
  __syncthreads();

  float vs = 0.f;
#pragma unroll
  for (int i = 0; i < 4; ++i) { float d = v[i] - mu; vs += d * d; }
  red[tid] = vs; __syncthreads();
  for (int sft = 128; sft > 0; sft >>= 1) {
    if (tid < sft) red[tid] += red[tid + sft];
    __syncthreads();
  }
  float rstd = rsqrtf(red[0] * (1.0f / D_) + EPS_);

#pragma unroll
  for (int i = 0; i < 4; ++i) {
    int c = tid + i * 256;
    float out = (v[i] - mu) * rstd * g[c] + beta[c];
    dstf[(size_t)row * D_ + c] = out;
    dsth[(size_t)row * D_ + c] = (_Float16)out;
  }
}

// ---------------------------------------------------------------------------
// Host orchestration
// ---------------------------------------------------------------------------
static inline size_t alignup(size_t v) { return (v + 255) & ~(size_t)255; }

extern "C" void kernel_launch(void* const* d_in, const int* in_sizes, int n_in,
                              void* d_out, int out_size, void* d_ws, size_t ws_size,
                              hipStream_t stream) {
  (void)in_sizes; (void)n_in; (void)out_size; (void)ws_size;

  const int*   seq    = (const int*)  d_in[0];
  const int*   mask   = (const int*)  d_in[1];
  const float* emb    = (const float*)d_in[2];
  const float* proj_w = (const float*)d_in[3];
  const float* proj_b = (const float*)d_in[4];
  const float* qw     = (const float*)d_in[5];
  const float* qb     = (const float*)d_in[6];
  const float* kw     = (const float*)d_in[7];
  const float* kb     = (const float*)d_in[8];
  const float* vw     = (const float*)d_in[9];
  const float* vb     = (const float*)d_in[10];
  const float* ow     = (const float*)d_in[11];
  const float* ob     = (const float*)d_in[12];
  const float* ln_g   = (const float*)d_in[13];
  const float* ln_b   = (const float*)d_in[14];

  char* base = (char*)d_ws;
  size_t off = 0;
  float*    x    = (float*)(base + off);    off = alignup(off + (size_t)NTOK * D_ * 4);
  float*    gbuf = (float*)(base + off);    off = alignup(off + (size_t)NTOK * D_ * 4);
  _Float16* x16  = (_Float16*)(base + off); off = alignup(off + (size_t)NTOK * D_ * 2);
  _Float16* ao16 = (_Float16*)(base + off); off = alignup(off + (size_t)NTOK * D_ * 2);
  _Float16* Q16  = (_Float16*)(base + off); off = alignup(off + (size_t)NTOK * D_ * 2);
  _Float16* K16  = (_Float16*)(base + off); off = alignup(off + (size_t)NTOK * D_ * 2);
  _Float16* V16  = (_Float16*)(base + off); off = alignup(off + (size_t)NTOK * D_ * 2);
  _Float16* xe16 = (_Float16*)(base + off); off = alignup(off + (size_t)NTOK * SEQDIM_ * 2);
  _Float16* WtP  = (_Float16*)(base + off); off = alignup(off + (size_t)SEQDIM_ * D_ * 2);
  _Float16* WtQ  = (_Float16*)(base + off); off = alignup(off + (size_t)L_ * D_ * D_ * 2);
  _Float16* WtK  = (_Float16*)(base + off); off = alignup(off + (size_t)L_ * D_ * D_ * 2);
  _Float16* WtV  = (_Float16*)(base + off); off = alignup(off + (size_t)L_ * D_ * D_ * 2);
  _Float16* WtO  = (_Float16*)(base + off); off = alignup(off + (size_t)L_ * D_ * D_ * 2);

  const float scale = 0.125f;   // 1/sqrt(64)
  const size_t DD = (size_t)D_ * D_;

  // --- one-time weight repack (f32 [K,N] -> f16 [N,K]) ---
  pack_wt<<<(SEQDIM_ * D_) / 256, 256, 0, stream>>>(proj_w, WtP, 7, D_);
  for (int l = 0; l < L_; ++l) {
    pack_wt<<<DD / 256, 256, 0, stream>>>(qw + l * DD, WtQ + l * DD, 10, D_);
    pack_wt<<<DD / 256, 256, 0, stream>>>(kw + l * DD, WtK + l * DD, 10, D_);
    pack_wt<<<DD / 256, 256, 0, stream>>>(vw + l * DD, WtV + l * DD, 10, D_);
    pack_wt<<<DD / 256, 256, 0, stream>>>(ow + l * DD, WtO + l * DD, 10, D_);
  }

  // --- embedding + input projection ---
  gather_embed<<<NTOK, 128, 0, stream>>>(seq, emb, xe16);
  {
    dim3 grid(NTOK / 128, D_ / 128);
    gemm_wmma<1><<<grid, 256, 0, stream>>>(xe16, WtP, proj_b, x, x16,
                                           NTOK, D_, SEQDIM_);
  }

  dim3 ggrid(NTOK / 128, D_ / 128);
  dim3 agrid(S_ / 64, B_ * H_);

  for (int l = 0; l < L_; ++l) {
    gemm_wmma<2><<<ggrid, 256, 0, stream>>>(x16, WtQ + l * DD, qb + l * D_,
                                            nullptr, Q16, NTOK, D_, D_);
    gemm_wmma<2><<<ggrid, 256, 0, stream>>>(x16, WtK + l * DD, kb + l * D_,
                                            nullptr, K16, NTOK, D_, D_);
    gemm_wmma<3><<<ggrid, 256, 0, stream>>>(x16, WtV + l * DD, vb + l * D_,
                                            nullptr, V16, NTOK, D_, D_);

    flash_attn<<<agrid, 128, 0, stream>>>(Q16, K16, V16, mask, ao16, scale);

    gemm_wmma<0><<<ggrid, 256, 0, stream>>>(ao16, WtO + l * DD, ob + l * D_,
                                            gbuf, nullptr, NTOK, D_, D_);

    float* dstf = (l == L_ - 1) ? (float*)d_out : x;
    add_layernorm<<<NTOK, 256, 0, stream>>>(x, gbuf, ln_g + l * D_,
                                            ln_b + l * D_, dstf, x16);
  }
}
